// HighResNet_1168231104531
// MI455X (gfx1250) — matrix operand order repaired and checked by reading, not compiled, not run on hardware
//
#include <hip/hip_runtime.h>
#include <hip/hip_bf16.h>
#include <stdint.h>

// ---------------------------------------------------------------------------
// DCRNN encoder/decoder for MI455X (gfx1250).
// All matmuls via v_wmma_f32_16x16x32_bf16 (bf16 in, fp32 accumulate).
// Diffusion GEMM stages its B-tile (S rows) in LDS, shared by 8 waves.
// ---------------------------------------------------------------------------

typedef __attribute__((ext_vector_type(16))) __bf16 v16bf;
typedef __attribute__((ext_vector_type(8)))  float  v8f;

union FragAB {
  v16bf    v;
  uint32_t u[8];
};

#define NN   512   // graph nodes
#define HD   64    // hidden dim
#define BB   64    // batch
#define TOTB 5     // diffusion blocks (identity + 2 supports * 2 steps)

#define SPITCH 520 // LDS row pitch in bf16 (260 dwords == 4 mod 64 -> conflict-free)

// ---------------------------------------------------------------------------
// elementwise / setup kernels
// ---------------------------------------------------------------------------
__global__ void cvt_bf16_k(const float* __restrict__ in, __bf16* __restrict__ out, int n) {
  int i = blockIdx.x * blockDim.x + threadIdx.x;
  if (i < n) out[i] = (__bf16)in[i];
}

// Wt[d*Cp + c] = W[c*D + d]  (transpose + convert, makes A rows K-contiguous)
__global__ void transpose_w_k(const float* __restrict__ W, __bf16* __restrict__ Wt,
                              int Cp, int D) {
  int i = blockIdx.x * blockDim.x + threadIdx.x;
  if (i >= Cp * D) return;
  int d = i / Cp, c = i % Cp;
  Wt[i] = (__bf16)W[(size_t)c * D + d];
}

// Build zf block0 = concat([x, (r?)*h]) in bf16, layout (B, C, N), n fastest.
__global__ void concat_z_k(const float* __restrict__ x, int xstride, int xoff, int Cx,
                           const float* __restrict__ h,      // (B, HD, NN)
                           const float* __restrict__ rgates, // (B, 2*HD, NN) or null
                           __bf16* __restrict__ z0, int C, int total) {
  int i = blockIdx.x * blockDim.x + threadIdx.x;
  if (i >= total) return;
  int n = i & (NN - 1);
  int c = (i >> 9) % C;
  int b = i / (C * NN);
  float v;
  if (c < Cx) {
    v = x[(size_t)(((b * Cx + c) << 9) + n) * xstride + xoff];
  } else {
    int hc = c - Cx;
    float hv = h[((size_t)(b * HD + hc) << 9) + n];
    if (rgates) hv *= rgates[((size_t)(b * 2 * HD + hc) << 9) + n];  // r * h
    v = hv;
  }
  z0[i] = (__bf16)v;
}

__global__ void sigmoid_k(float* __restrict__ g, int total) {
  int i = blockIdx.x * blockDim.x + threadIdx.x;
  if (i < total) g[i] = 1.0f / (1.0f + expf(-g[i]));
}

// h = u*h + (1-u)*tanh(cand) ; u = gates[:, HD:2HD, :]
__global__ void combine_h_k(float* __restrict__ h, const float* __restrict__ gates,
                            const float* __restrict__ cand, int total) {
  int i = blockIdx.x * blockDim.x + threadIdx.x;
  if (i >= total) return;
  int n  = i & (NN - 1);
  int hc = (i >> 9) & (HD - 1);
  int b  = i >> 15;                   // /(HD*NN)
  float u = gates[((size_t)(b * 2 * HD + HD + hc) << 9) + n];
  float c = tanhf(cand[i]);
  h[i] = u * h[i] + (1.0f - u) * c;
}

// y[b,n] = proj_b + sum_h proj_w[h]*h1[b,h,n]; write to out[...,t] and xdec
__global__ void proj_k(const float* __restrict__ h1, const float* __restrict__ pw,
                       const float* __restrict__ pb, float* __restrict__ out,
                       float* __restrict__ xdec, int t) {
  int i = blockIdx.x * blockDim.x + threadIdx.x;   // B*N
  if (i >= BB * NN) return;
  int n = i & (NN - 1), b = i >> 9;
  const float* hb = h1 + ((size_t)b * HD << 9) + n;
  float acc = pb[0];
#pragma unroll
  for (int k = 0; k < HD; ++k) acc += pw[k] * hb[(size_t)k << 9];
  out[(size_t)i * 12 + t] = acc;   // (B,1,N,HOR)
  xdec[i] = acc;
}

// ---------------------------------------------------------------------------
// Diffusion GEMM:  Y[r,n] = sum_m X[r,m] * S[n,m]   (M x 512) @ S^T
// 256-thread workgroup = 8 waves covering 128 M-rows; the 16 S-rows that form
// the B operand for this column tile (a contiguous 16 KB block) are staged in
// LDS once and reused by all 8 waves (8x cut in L2 traffic for S).
// ---------------------------------------------------------------------------
__global__ __launch_bounds__(256)
void diff_gemm_k(const __bf16* __restrict__ X,   // (M, 512) bf16 row-major
                 const __bf16* __restrict__ S,   // (512, 512) bf16 row-major
                 __bf16* __restrict__ Y,         // (M, 512) bf16
                 int M) {
  __shared__ __bf16 sB[16 * SPITCH];             // padded pitch, ~16.3 KB
  const int tid = threadIdx.x;
  const int tn  = blockIdx.x;                    // N tile (0..31)

  // stage B tile: S rows [tn*16, tn*16+16) are contiguous (1024 uint4)
  {
    const uint4* src = (const uint4*)(S + ((size_t)tn * 16 << 9));
    uint4*       dst = (uint4*)sB;
#pragma unroll
    for (int p = 0; p < 4; ++p) {
      int idx = p * 256 + tid;                   // 0..1023
      int r = idx >> 6, c = idx & 63;            // 64 uint4 per source row
      dst[r * (SPITCH / 8) + c] = src[idx];      // SPITCH/8 = 65 uint4 pitch
    }
  }
  __syncthreads();

  const int wave    = tid >> 5;
  const int lane    = tid & 31;
  const int rowbase = blockIdx.y * 128 + wave * 16;
  if (rowbase >= M) return;                      // tail block: wave-uniform exit

  const int m16    = lane & 15;
  const int laneHi = lane >> 4;                  // 0 or 1
  const int khA    = laneHi * 16;                // A: upper lanes hold K=16..31
  const int kbB    = laneHi * 16;                // B: upper lanes hold K=16..31
  const __bf16* Arow = X + ((size_t)(rowbase + m16) << 9);
  const __bf16* Brow = sB + m16 * SPITCH;        // column n of B = staged row

  v8f acc = {0.f, 0.f, 0.f, 0.f, 0.f, 0.f, 0.f, 0.f};
  for (int kk = 0; kk < NN; kk += 32) {
    if (kk + 64 < NN) __builtin_prefetch(Arow + kk + 64, 0, 1);
    FragAB a, b;
#pragma unroll
    for (int j = 0; j < 8; ++j) {
      int ka = kk + 2 * j + khA;
      a.u[j] = *(const uint32_t*)(Arow + ka);
      int kb = kk + kbB + 2 * j;
      b.u[j] = *(const uint32_t*)(Brow + kb);
    }
    acc = __builtin_amdgcn_wmma_f32_16x16x32_bf16(false, a.v, false, b.v,
                                                  (short)0, acc, false, false);
  }
#pragma unroll
  for (int j = 0; j < 8; ++j) {
    int r = rowbase + j + laneHi * 8;            // C/D: VGPR j -> row j / j+8
    Y[((size_t)r << 9) + tn * 16 + m16] = (__bf16)acc[j];
  }
}

// ---------------------------------------------------------------------------
// Weight GEMM: out[b,d,n] = bias[d] + sum_{t,c} Wt[d, t*C+c] * zf[t,b,c,n]
// zf layout (TOT, B, C, N); K zero-padded per 32-block (C not mult of 32).
// ---------------------------------------------------------------------------
__global__ __launch_bounds__(32)
void wgt_gemm_k(const __bf16* __restrict__ Wt,   // (D, 5*C) bf16 row-major
                const __bf16* __restrict__ zf,   // (5, B, C, N) bf16
                const float* __restrict__ bias,  // (D)
                float* __restrict__ out,         // (B, D, N) f32
                int C, int D) {
  const int lane   = threadIdx.x;
  const int tn     = blockIdx.x;          // 0..31
  const int tm     = blockIdx.y;          // 0..D/16-1
  const int bidx   = blockIdx.z;          // batch
  const int Cp     = TOTB * C;
  const int m16    = lane & 15;
  const int laneHi = lane >> 4;
  const int khA    = laneHi * 16;
  const int kbB    = laneHi * 16;
  const int n      = tn * 16 + m16;
  const __bf16* Arow = Wt + (size_t)(tm * 16 + m16) * Cp;
  v8f acc = {0.f, 0.f, 0.f, 0.f, 0.f, 0.f, 0.f, 0.f};
  for (int t = 0; t < TOTB; ++t) {
    const __bf16* Bblk = zf + (((size_t)t * BB + bidx) * C << 9);
    const __bf16* Ablk = Arow + t * C;
    for (int k0 = 0; k0 < C; k0 += 32) {
      FragAB a, bf;
#pragma unroll
      for (int j = 0; j < 8; ++j) {
        int ka = k0 + 2 * j + khA;
        uint32_t alo = (ka     < C) ? (uint32_t)*(const unsigned short*)(Ablk + ka)     : 0u;
        uint32_t ahi = (ka + 1 < C) ? (uint32_t)*(const unsigned short*)(Ablk + ka + 1) : 0u;
        a.u[j] = alo | (ahi << 16);
        int kb = k0 + kbB + 2 * j;
        uint32_t blo = (kb     < C) ? (uint32_t)*(const unsigned short*)(Bblk + ((size_t)kb << 9) + n)       : 0u;
        uint32_t bhi = (kb + 1 < C) ? (uint32_t)*(const unsigned short*)(Bblk + ((size_t)(kb + 1) << 9) + n) : 0u;
        bf.u[j] = blo | (bhi << 16);
      }
      acc = __builtin_amdgcn_wmma_f32_16x16x32_bf16(false, a.v, false, bf.v,
                                                    (short)0, acc, false, false);
    }
  }
#pragma unroll
  for (int j = 0; j < 8; ++j) {
    int d = tm * 16 + j + laneHi * 8;
    out[(((size_t)bidx * D + d) << 9) + n] = acc[j] + bias[d];
  }
}

// ---------------------------------------------------------------------------
// host orchestration
// ---------------------------------------------------------------------------
extern "C" void kernel_launch(void* const* d_in, const int* in_sizes, int n_in,
                              void* d_out, int out_size, void* d_ws, size_t ws_size,
                              hipStream_t stream) {
  (void)in_sizes; (void)n_in; (void)out_size; (void)ws_size;
  const float* inputs   = (const float*)d_in[0];   // (B,2,N,SEQ)
  const float* supports = (const float*)d_in[1];   // (2,N,N)
  const float* wgs[4] = {(const float*)d_in[2],  (const float*)d_in[6],
                         (const float*)d_in[10], (const float*)d_in[14]};
  const float* bgs[4] = {(const float*)d_in[3],  (const float*)d_in[7],
                         (const float*)d_in[11], (const float*)d_in[15]};
  const float* wcs[4] = {(const float*)d_in[4],  (const float*)d_in[8],
                         (const float*)d_in[12], (const float*)d_in[16]};
  const float* bcs[4] = {(const float*)d_in[5],  (const float*)d_in[9],
                         (const float*)d_in[13], (const float*)d_in[17]};
  const float* proj_w = (const float*)d_in[18];
  const float* proj_b = (const float*)d_in[19];
  float* out = (float*)d_out;

  // workspace carving (256B aligned)
  char* wsp = (char*)d_ws;
  auto carve = [&](size_t bytes) -> void* {
    void* p = (void*)wsp;
    wsp += (bytes + 255) & ~(size_t)255;
    return p;
  };
  const int Cps[4] = {5 * (2 + HD), 5 * (HD + HD), 5 * (1 + HD), 5 * (HD + HD)}; // 330,640,325,640
  __bf16* Sbf = (__bf16*)carve((size_t)2 * NN * NN * 2);
  __bf16 *WgT[4], *WcT[4];
  for (int i = 0; i < 4; ++i) {
    WgT[i] = (__bf16*)carve((size_t)Cps[i] * 2 * HD * 2);
    WcT[i] = (__bf16*)carve((size_t)Cps[i] * HD * 2);
  }
  __bf16* zf   = (__bf16*)carve((size_t)TOTB * BB * 128 * NN * 2);
  float* gates = (float*)carve((size_t)BB * 2 * HD * NN * 4);
  float* cand  = (float*)carve((size_t)BB * HD * NN * 4);
  float* h0    = (float*)carve((size_t)BB * HD * NN * 4);
  float* h1    = (float*)carve((size_t)BB * HD * NN * 4);
  float* xdec  = (float*)carve((size_t)BB * NN * 4);

  // init states + decoder go-token, convert supports + weights once (deterministic)
  hipMemsetAsync(h0,   0, (size_t)BB * HD * NN * 4, stream);
  hipMemsetAsync(h1,   0, (size_t)BB * HD * NN * 4, stream);
  hipMemsetAsync(xdec, 0, (size_t)BB * NN * 4, stream);
  cvt_bf16_k<<<(2 * NN * NN + 255) / 256, 256, 0, stream>>>(supports, Sbf, 2 * NN * NN);
  for (int i = 0; i < 4; ++i) {
    int tg = Cps[i] * 2 * HD;
    transpose_w_k<<<(tg + 255) / 256, 256, 0, stream>>>(wgs[i], WgT[i], Cps[i], 2 * HD);
    int tc = Cps[i] * HD;
    transpose_w_k<<<(tc + 255) / 256, 256, 0, stream>>>(wcs[i], WcT[i], Cps[i], HD);
  }

  auto run_diff = [&](int C) {
    int M = BB * C;
    size_t blk = (size_t)M * NN;
    dim3 grid(NN / 16, (M + 127) / 128);
    diff_gemm_k<<<grid, 256, 0, stream>>>(zf,           Sbf,           zf + 1 * blk, M);
    diff_gemm_k<<<grid, 256, 0, stream>>>(zf + 1 * blk, Sbf,           zf + 2 * blk, M);
    diff_gemm_k<<<grid, 256, 0, stream>>>(zf,           Sbf + NN * NN, zf + 3 * blk, M);
    diff_gemm_k<<<grid, 256, 0, stream>>>(zf + 3 * blk, Sbf + NN * NN, zf + 4 * blk, M);
  };

  auto cell = [&](const float* x, int xstride, int xoff, int Cx, float* h, int wi) {
    const int C = Cx + HD;
    const int totZ = BB * C * NN;
    // gate path: ru = sigmoid(gconv(x, h))
    concat_z_k<<<(totZ + 255) / 256, 256, 0, stream>>>(x, xstride, xoff, Cx, h, nullptr, zf, C, totZ);
    run_diff(C);
    { dim3 g(NN / 16, (2 * HD) / 16, BB);
      wgt_gemm_k<<<g, 32, 0, stream>>>(WgT[wi], zf, bgs[wi], gates, C, 2 * HD); }
    { int tot = BB * 2 * HD * NN;
      sigmoid_k<<<(tot + 255) / 256, 256, 0, stream>>>(gates, tot); }
    // candidate path: c = tanh(gconv(x, r*h)); h = u*h + (1-u)*c
    concat_z_k<<<(totZ + 255) / 256, 256, 0, stream>>>(x, xstride, xoff, Cx, h, gates, zf, C, totZ);
    run_diff(C);
    { dim3 g(NN / 16, HD / 16, BB);
      wgt_gemm_k<<<g, 32, 0, stream>>>(WcT[wi], zf, bcs[wi], cand, C, HD); }
    { int tot = BB * HD * NN;
      combine_h_k<<<(tot + 255) / 256, 256, 0, stream>>>(h, gates, cand, tot); }
  };

  // encoder: SEQ=12 steps, x strided (B,2,N,SEQ)
  for (int t = 0; t < 12; ++t) {
    cell(inputs, 12, t, 2,  h0, 0);
    cell(h0,      1, 0, HD, h1, 1);
  }
  // decoder: HOR=12 steps
  for (int t = 0; t < 12; ++t) {
    cell(xdec, 1, 0, 1,  h0, 2);
    cell(h0,   1, 0, HD, h1, 3);
    proj_k<<<(BB * NN + 255) / 256, 256, 0, stream>>>(h1, proj_w, proj_b, out, xdec, t);
  }
}
